// SNNParallelMSO_678604833398
// MI455X (gfx1250) — matrix operand order repaired and checked by reading, not compiled, not run on hardware
//
#include <hip/hip_runtime.h>
#include <stdint.h>

// ---------------------------------------------------------------------------
// SNN forward (snntorch-style LIF) for MI455X / gfx1250, wave32 + WMMA bf16.
// Per step:  cur1 = x_t @ Wcat^T + bcat  (256x512x2048)
//            cur2 = spk1 @ W2^T + b2     (256x2048x1024)
// GEMM: block tile 32x128, K-chunk 64, double-buffered LDS, async-to-LDS
// staging when the toolchain exposes it, 2 WMMAs per A-fragment per wave.
// ---------------------------------------------------------------------------

typedef __attribute__((ext_vector_type(8)))  __bf16 v8bf;
typedef __attribute__((ext_vector_type(16))) __bf16 v16bf;
typedef __attribute__((ext_vector_type(8)))  float  v8f;
typedef __attribute__((__vector_size__(16))) int    v4i32;   // builtin's V4i

#define T_STEPS 256
#define BATCH   256
#define IN_F    512
#define H1      2048      // fast (1024) + slow (1024) concatenated
#define H2      1024
#define BK      64
#define THR     1.0f
#define BETA_FAST 0.60653065971263342f   // exp(-1/2)
#define BETA_SLOW 0.95122942450071400f   // exp(-1/20)
#define BETA_IC   0.90483741803595952f   // exp(-1/10)

#if defined(__has_builtin)
#  if __has_builtin(__builtin_amdgcn_global_load_async_to_lds_b128)
#    define USE_ASYNC_LDS 1
#  endif
#endif
#ifndef USE_ASYNC_LDS
#  define USE_ASYNC_LDS 0
#endif

typedef __attribute__((address_space(1))) v4i32 gv4i;
typedef __attribute__((address_space(3))) v4i32 lv4i;

__device__ __forceinline__ void async_cp16(const void* g, void* l) {
#if USE_ASYNC_LDS
    __builtin_amdgcn_global_load_async_to_lds_b128(
        (gv4i*)(uintptr_t)g, (lv4i*)(unsigned int)(uintptr_t)l, 0, 0);
#endif
}
__device__ __forceinline__ void async_wait0() {
#if USE_ASYNC_LDS
#  if __has_builtin(__builtin_amdgcn_s_wait_asynccnt)
    __builtin_amdgcn_s_wait_asynccnt(0);
#  else
    asm volatile("s_wait_asynccnt 0x0" ::: "memory");
#  endif
#endif
}

__device__ __forceinline__ unsigned short f2bf(float f) {
    union { float f; unsigned int u; } v; v.f = f;
    unsigned int u = v.u;
    unsigned int r = u + 0x7FFFu + ((u >> 16) & 1u);   // round-to-nearest-even
    return (unsigned short)(r >> 16);
}

__device__ __forceinline__ v16bf cat8(v8bf lo, v8bf hi) {
    return __builtin_shufflevector(lo, hi, 0,1,2,3,4,5,6,7,8,9,10,11,12,13,14,15);
}

// ---------------------------------------------------------------------------
// Shared 32x128 GEMM core: C(32x128 f32) += A(32xK bf16) * B(128xK bf16)^T.
// 8 waves: wave tile 16(M) x 32(N) => accumulators c0 (n0) and c1 (n0+16).
// LDS: As[2][32*64], Bs[2][128*64] (double buffered).
__device__ __forceinline__ void wmma_gemm_32x128(
    const unsigned short* __restrict__ Ag, int aStride,
    const unsigned short* __restrict__ Bg, int bStride, int K,
    int gM, int gN,
    unsigned short* __restrict__ As, unsigned short* __restrict__ Bs,
    v8f& c0, v8f& c1)
{
    const int tid  = threadIdx.x;
    const int lane = tid & 31;
    const int wave = tid >> 5;
    const int half = lane >> 4;
    const int l15  = lane & 15;
    const int m0 = (wave >> 2) * 16;
    const int n0 = (wave & 3) * 32;

    const int arow = tid >> 3, acol = (tid & 7) * 8;   // A stage: 16B/thread
    const int brow = tid >> 1, bcol = (tid & 1) * 32;  // B stage: 64B/thread

    const unsigned short* agp = Ag + (size_t)(gM + arow) * aStride + acol;
    const unsigned short* bgp = Bg + (size_t)(gN + brow) * bStride + bcol;
    unsigned short* asp = As + arow * BK + acol;
    unsigned short* bsp = Bs + brow * BK + bcol;

    const int ABUF = 32 * BK;    // elements per A buffer
    const int BBUF = 128 * BK;   // elements per B buffer
    const int nchunks = K / BK;

#if USE_ASYNC_LDS
    // Prologue: async-stage chunk 0 into buffer 0.
    async_cp16(agp, asp);
    async_cp16(bgp +  0, bsp +  0);
    async_cp16(bgp +  8, bsp +  8);
    async_cp16(bgp + 16, bsp + 16);
    async_cp16(bgp + 24, bsp + 24);
    async_wait0();
    __syncthreads();
    for (int ch = 0; ch < nchunks; ++ch) {
        const int p = ch & 1;
        if (ch + 1 < nchunks) {   // fire-and-forget stage of next chunk
            const unsigned short* ag = agp + (size_t)(ch + 1) * BK;
            const unsigned short* bg = bgp + (size_t)(ch + 1) * BK;
            unsigned short* as = asp + (p ^ 1) * ABUF;
            unsigned short* bs = bsp + (p ^ 1) * BBUF;
            async_cp16(ag, as);
            async_cp16(bg +  0, bs +  0);
            async_cp16(bg +  8, bs +  8);
            async_cp16(bg + 16, bs + 16);
            async_cp16(bg + 24, bs + 24);
        }
        const unsigned short* Ab = As + p * ABUF;
        const unsigned short* Bb = Bs + p * BBUF;
#pragma unroll
        for (int s = 0; s < 2; ++s) {
            const int ks = s * 32;
            const unsigned short* ar = Ab + (m0 + l15) * BK + ks + half * 16;
            v16bf a = cat8(*(const v8bf*)ar, *(const v8bf*)(ar + 8));
            const unsigned short* br0 = Bb + (n0 + l15) * BK + ks + half * 16;
            const unsigned short* br1 = br0 + 16 * BK;
            v16bf b0 = cat8(*(const v8bf*)br0, *(const v8bf*)(br0 + 8));
            v16bf b1 = cat8(*(const v8bf*)br1, *(const v8bf*)(br1 + 8));
            c0 = __builtin_amdgcn_wmma_f32_16x16x32_bf16(false, a, false, b0,
                                                         (short)0, c0, false, false);
            c1 = __builtin_amdgcn_wmma_f32_16x16x32_bf16(false, a, false, b1,
                                                         (short)0, c1, false, false);
        }
        async_wait0();
        __syncthreads();
    }
#else
    // Register double-buffer fallback.
    uint4 ra, rb0, rb1, rb2, rb3;
    ra  = *(const uint4*)agp;
    rb0 = *(const uint4*)(bgp + 0);
    rb1 = *(const uint4*)(bgp + 8);
    rb2 = *(const uint4*)(bgp + 16);
    rb3 = *(const uint4*)(bgp + 24);
    *(uint4*)asp = ra;
    *(uint4*)(bsp +  0) = rb0;
    *(uint4*)(bsp +  8) = rb1;
    *(uint4*)(bsp + 16) = rb2;
    *(uint4*)(bsp + 24) = rb3;
    __syncthreads();
    for (int ch = 0; ch < nchunks; ++ch) {
        const int p = ch & 1;
        const bool more = (ch + 1 < nchunks);
        if (more) {   // global loads in flight during compute
            const unsigned short* ag = agp + (size_t)(ch + 1) * BK;
            const unsigned short* bg = bgp + (size_t)(ch + 1) * BK;
            __builtin_prefetch(bg + BK, 0, 0);
            ra  = *(const uint4*)ag;
            rb0 = *(const uint4*)(bg + 0);
            rb1 = *(const uint4*)(bg + 8);
            rb2 = *(const uint4*)(bg + 16);
            rb3 = *(const uint4*)(bg + 24);
        }
        const unsigned short* Ab = As + p * ABUF;
        const unsigned short* Bb = Bs + p * BBUF;
#pragma unroll
        for (int s = 0; s < 2; ++s) {
            const int ks = s * 32;
            const unsigned short* ar = Ab + (m0 + l15) * BK + ks + half * 16;
            v16bf a = cat8(*(const v8bf*)ar, *(const v8bf*)(ar + 8));
            const unsigned short* br0 = Bb + (n0 + l15) * BK + ks + half * 16;
            const unsigned short* br1 = br0 + 16 * BK;
            v16bf b0 = cat8(*(const v8bf*)br0, *(const v8bf*)(br0 + 8));
            v16bf b1 = cat8(*(const v8bf*)br1, *(const v8bf*)(br1 + 8));
            c0 = __builtin_amdgcn_wmma_f32_16x16x32_bf16(false, a, false, b0,
                                                         (short)0, c0, false, false);
            c1 = __builtin_amdgcn_wmma_f32_16x16x32_bf16(false, a, false, b1,
                                                         (short)0, c1, false, false);
        }
        if (more) {
            unsigned short* as = asp + (p ^ 1) * ABUF;
            unsigned short* bs = bsp + (p ^ 1) * BBUF;
            *(uint4*)as = ra;
            *(uint4*)(bs +  0) = rb0;
            *(uint4*)(bs +  8) = rb1;
            *(uint4*)(bs + 16) = rb2;
            *(uint4*)(bs + 24) = rb3;
        }
        __syncthreads();
    }
#endif
}

// ---------------------------------------------------------------------------
__global__ void zero_f32(float* __restrict__ p, int n) {
    int i = blockIdx.x * blockDim.x + threadIdx.x;
    int stride = gridDim.x * blockDim.x;
    for (; i < n; i += stride) p[i] = 0.0f;
}

// Convert Wf|Ws -> Wcat (bf16, [2048][512]) and W2 -> W2b (bf16, [1024][2048])
__global__ void convert_weights(const float* __restrict__ Wf,
                                const float* __restrict__ Ws,
                                const float* __restrict__ W2,
                                unsigned short* __restrict__ Wcat,
                                unsigned short* __restrict__ W2b) {
    const int n1 = H1 * IN_F;
    const int n2 = H2 * H1;
    int i = blockIdx.x * blockDim.x + threadIdx.x;
    int stride = gridDim.x * blockDim.x;
    for (int k = i; k < n1 + n2; k += stride) {
        if (k < n1) {
            int row = k / IN_F, col = k % IN_F;
            float v = (row < 1024) ? Wf[row * IN_F + col]
                                   : Ws[(row - 1024) * IN_F + col];
            Wcat[k] = f2bf(v);
        } else {
            int k2 = k - n1;
            W2b[k2] = f2bf(W2[k2]);
        }
    }
}

// Per-step x_t (f32) -> bf16, done once (not per N-block). 4 elems/thread.
__global__ __launch_bounds__(256) void conv_x(const float* __restrict__ xt,
                                              unsigned short* __restrict__ xb,
                                              int n4) {
    int i = blockIdx.x * blockDim.x + threadIdx.x;
    if (i < n4) {
        float4 v = ((const float4*)xt)[i];
        unsigned int lo = (unsigned int)f2bf(v.x) | ((unsigned int)f2bf(v.y) << 16);
        unsigned int hi = (unsigned int)f2bf(v.z) | ((unsigned int)f2bf(v.w) << 16);
        ((uint2*)xb)[i] = make_uint2(lo, hi);
    }
}

// ---------------------------------------------------------------------------
// Layer 1: cur = x_t @ Wcat^T + bias; LIF per column-group; emit spk1 (bf16).
__global__ __launch_bounds__(256) void snn_layer1(
    const unsigned short* __restrict__ xb, const unsigned short* __restrict__ Wcat,
    const float* __restrict__ bfp, const float* __restrict__ bsp,
    float* __restrict__ mem1, unsigned short* __restrict__ spk1,
    float* __restrict__ s1_accum)
{
    __shared__ __align__(16) unsigned short As[2][32 * BK];
    __shared__ __align__(16) unsigned short Bs[2][128 * BK];
    __shared__ float red[256];

    const int gM = blockIdx.x * 32;
    const int gN = blockIdx.y * 128;
    v8f c0 = {}, c1 = {};
    wmma_gemm_32x128(xb, IN_F, Wcat, IN_F, IN_F, gM, gN, &As[0][0], &Bs[0][0], c0, c1);

    const int tid  = threadIdx.x;
    const int lane = tid & 31;
    const int wave = tid >> 5;
    const int half = lane >> 4;
    const int l15  = lane & 15;
    const int m0 = (wave >> 2) * 16;
    const int n0 = (wave & 3) * 32;

    float cnt = 0.0f;
#pragma unroll
    for (int ti = 0; ti < 2; ++ti) {
        v8f c = ti ? c1 : c0;
        const int gcol = gN + n0 + ti * 16 + l15;
        const float bias = (gcol < 1024) ? bfp[gcol] : bsp[gcol - 1024];
        const float beta = (gcol < 1024) ? BETA_FAST : BETA_SLOW;
#pragma unroll
        for (int r = 0; r < 8; ++r) {
            const int grow = gM + m0 + r + half * 8;   // C layout: M = r + half*8
            const size_t idx = (size_t)grow * H1 + gcol;
            float cur   = c[r] + bias;
            float mem   = mem1[idx];
            float reset = (mem > THR) ? 1.0f : 0.0f;
            float mnew  = beta * mem + cur - reset * THR;
            float spk   = (mnew > THR) ? 1.0f : 0.0f;
            mem1[idx] = mnew;
            spk1[idx] = f2bf(spk);
            cnt += spk;
        }
    }
    red[tid] = cnt;
    __syncthreads();
    for (int s = 128; s > 0; s >>= 1) {
        if (tid < s) red[tid] += red[tid + s];
        __syncthreads();
    }
    if (tid == 0) atomicAdd(s1_accum, red[0]);
}

// ---------------------------------------------------------------------------
// Layer 2: cur2 = spk1 @ W2^T + b2; LIF (beta_ic); mem2_sum += mem2.
__global__ __launch_bounds__(256) void snn_layer2(
    const unsigned short* __restrict__ spk1, const unsigned short* __restrict__ W2b,
    const float* __restrict__ b2, float* __restrict__ mem2,
    float* __restrict__ m2sum, float* __restrict__ s2_accum)
{
    __shared__ __align__(16) unsigned short As[2][32 * BK];
    __shared__ __align__(16) unsigned short Bs[2][128 * BK];
    __shared__ float red[256];

    const int gM = blockIdx.x * 32;
    const int gN = blockIdx.y * 128;
    v8f c0 = {}, c1 = {};
    wmma_gemm_32x128(spk1, H1, W2b, H1, H1, gM, gN, &As[0][0], &Bs[0][0], c0, c1);

    const int tid  = threadIdx.x;
    const int lane = tid & 31;
    const int wave = tid >> 5;
    const int half = lane >> 4;
    const int l15  = lane & 15;
    const int m0 = (wave >> 2) * 16;
    const int n0 = (wave & 3) * 32;

    float cnt = 0.0f;
#pragma unroll
    for (int ti = 0; ti < 2; ++ti) {
        v8f c = ti ? c1 : c0;
        const int gcol = gN + n0 + ti * 16 + l15;
        const float bias = b2[gcol];
#pragma unroll
        for (int r = 0; r < 8; ++r) {
            const int grow = gM + m0 + r + half * 8;
            const size_t idx = (size_t)grow * H2 + gcol;
            float cur   = c[r] + bias;
            float mem   = mem2[idx];
            float reset = (mem > THR) ? 1.0f : 0.0f;
            float mnew  = BETA_IC * mem + cur - reset * THR;
            float spk   = (mnew > THR) ? 1.0f : 0.0f;
            mem2[idx]  = mnew;
            m2sum[idx] = m2sum[idx] + mnew;
            cnt += spk;
        }
    }
    red[tid] = cnt;
    __syncthreads();
    for (int s = 128; s > 0; s >>= 1) {
        if (tid < s) red[tid] += red[tid + s];
        __syncthreads();
    }
    if (tid == 0) atomicAdd(s2_accum, red[0]);
}

// ---------------------------------------------------------------------------
// Readout: out[256][2] = (m2sum/T) @ Wr^T + br ; plus spike means.
__global__ __launch_bounds__(512) void snn_readout(
    const float* __restrict__ m2sum, const float* __restrict__ Wr,
    const float* __restrict__ br, const float* __restrict__ scal,
    float* __restrict__ out)
{
    int tid = blockIdx.x * blockDim.x + threadIdx.x;
    if (tid < 512) {
        int row = tid >> 1, col = tid & 1;
        const float* mrow = m2sum + (size_t)row * H2;
        const float* wrow = Wr + (size_t)col * H2;
        float acc = 0.0f;
        for (int k = 0; k < H2; ++k) acc += mrow[k] * wrow[k];
        out[tid] = acc * (1.0f / (float)T_STEPS) + br[col];
    }
    if (tid == 0) {
        out[512] = scal[0] / ((float)T_STEPS * (float)BATCH * (float)H1);
        out[513] = scal[1] / ((float)T_STEPS * (float)BATCH * (float)H2);
    }
}

// ---------------------------------------------------------------------------
extern "C" void kernel_launch(void* const* d_in, const int* in_sizes, int n_in,
                              void* d_out, int out_size, void* d_ws, size_t ws_size,
                              hipStream_t stream)
{
    const float* spikes = (const float*)d_in[0];
    const float* Wf = (const float*)d_in[1];
    const float* bf = (const float*)d_in[2];
    const float* Ws = (const float*)d_in[3];
    const float* bs = (const float*)d_in[4];
    const float* W2 = (const float*)d_in[5];
    const float* b2 = (const float*)d_in[6];
    const float* Wr = (const float*)d_in[7];
    const float* br = (const float*)d_in[8];
    float* out = (float*)d_out;

    // Workspace layout (bytes):
    //   [0,          2,097,152)  Wcat bf16 [2048][512]
    //   [2,097,152,  6,291,456)  W2b  bf16 [1024][2048]
    //   [6,291,456,  8,388,608)  mem1 f32  [256][2048]
    //   [8,388,608,  9,437,184)  mem2 f32  [256][1024]
    //   [9,437,184, 10,485,760)  m2sum f32 [256][1024]
    //   [10,485,760,11,534,336)  spk1 bf16 [256][2048]
    //   [11,534,336,11,796,480)  xb   bf16 [256][512]
    //   [11,796,480,11,796,488)  scal f32  {s1, s2}
    char* ws = (char*)d_ws;
    unsigned short* Wcat = (unsigned short*)(ws + 0);
    unsigned short* W2b  = (unsigned short*)(ws + 2097152);
    float* mem1  = (float*)(ws + 6291456);
    float* mem2  = (float*)(ws + 8388608);
    float* m2sum = (float*)(ws + 9437184);
    unsigned short* spk1 = (unsigned short*)(ws + 10485760);
    unsigned short* xb   = (unsigned short*)(ws + 11534336);
    float* scal  = (float*)(ws + 11796480);

    // Zero all mutable state (mem1, mem2, m2sum, spk1, xb, scalars).
    {
        int nz = (11796488 - 6291456) / 4;
        zero_f32<<<512, 256, 0, stream>>>((float*)(ws + 6291456), nz);
    }
    convert_weights<<<1024, 256, 0, stream>>>(Wf, Ws, W2, Wcat, W2b);

    for (int t = 0; t < T_STEPS; ++t) {
        conv_x<<<128, 256, 0, stream>>>(spikes + (size_t)t * BATCH * IN_F, xb,
                                        BATCH * IN_F / 4);
        snn_layer1<<<dim3(8, 16), 256, 0, stream>>>(xb, Wcat, bf, bs,
                                                    mem1, spk1, scal);
        snn_layer2<<<dim3(8, 8), 256, 0, stream>>>(spk1, W2b, b2,
                                                   mem2, m2sum, scal + 1);
    }
    snn_readout<<<1, 512, 0, stream>>>(m2sum, Wr, br, scal, out);
}
